// HopfieldAttentionWrapper_18571438588244
// MI455X (gfx1250) — matrix operand order repaired and checked by reading, not compiled, not run on hardware
//
#include <hip/hip_runtime.h>
#include <hip/hip_bf16.h>
#include <math.h>

// ---------------------------------------------------------------------------
// Problem constants (reference): B=2, S=1024, D=2048, H=16, KV=8, HD=128
// ---------------------------------------------------------------------------
#define B_    2
#define S_    1024
#define D_    2048
#define H_    16
#define KV_   8
#define HD_   128
#define EQ_   2048          // H*HD
#define EK_   1024          // KV*HD
#define ETOT_ 4096          // EQ + 2*EK
#define M_    2048          // B*S
#define SCALING_ 0.08838834764831845f  // HD^-0.5
#define NEGBIG_ -1.0e9f

typedef __bf16 bh;
typedef __attribute__((ext_vector_type(16))) __bf16 v16bf;
typedef __attribute__((ext_vector_type(8)))  float  v8f;

// WMMA lane layouts used in this file (wave32; A/B k order is this code's choice, not the ISA table):
//   A (16x32 bf16): M = lane&15, element e -> K = (lane>>4)*16 + e (contiguous)
//   B (32x16 bf16): N = lane&15, element e -> K = (lane>>4)*16 + e (contiguous)
//   C/D (16x16 f32): reg r -> M = r + 8*(lane>>4), N = lane&15

// ---------------------------------------------------------------------------
// Kernel 1: QKV projection. QKV(f32, 2048 x 4096) = X @ [Wq|Wk|Wv]^T
// 64x64 workgroup tile, LDS-staged bf16 panels, 4 waves x (16x64) strips.
// The 64-wide N-block never straddles the Wq/Wk/Wv boundaries (2048, 3072),
// so the weight-matrix select is uniform per workgroup (scalar branch only).
// ---------------------------------------------------------------------------
__global__ void __launch_bounds__(128) k_qkv(const float* __restrict__ X,
                                             const float* __restrict__ Wq,
                                             const float* __restrict__ Wk,
                                             const float* __restrict__ Wv,
                                             float* __restrict__ QKV)
{
  __shared__ __align__(32) bh Ab[64 * 32];   // [m][k]
  __shared__ __align__(32) bh Bb[64 * 32];   // [n][k]
  const int tid  = threadIdx.x;
  const int wave = tid >> 5;
  const int lane = tid & 31;
  const int lh = lane >> 4, ln = lane & 15;
  const int bmi = blockIdx.x >> 6;           // 32 M-blocks
  const int bni = blockIdx.x & 63;           // 64 N-blocks
  const int m0 = bmi * 64, n0 = bni * 64;
  const int khalf = lh * 16;

  // uniform (per-workgroup) weight base select
  const float* Wbase;
  int e0;
  if (n0 < EQ_)            { Wbase = Wq; e0 = n0; }
  else if (n0 < EQ_ + EK_) { Wbase = Wk; e0 = n0 - EQ_; }
  else                     { Wbase = Wv; e0 = n0 - EQ_ - EK_; }

  v8f acc[4] = {};

  for (int kb = 0; kb < D_ / 32; ++kb) {
    const int kbase = kb * 32;
    // cooperative load + fp32->bf16 convert (b128 global loads)
#pragma unroll
    for (int i = 0; i < 4; ++i) {
      const int c  = tid + i * 128;          // 0..511 float4 chunks
      const int rr = c >> 3;
      const int k4 = (c & 7) * 4;
      const float4 xa =
          *(const float4*)&X[(size_t)(m0 + rr) * D_ + kbase + k4];
      Ab[rr * 32 + k4 + 0] = (bh)xa.x;  Ab[rr * 32 + k4 + 1] = (bh)xa.y;
      Ab[rr * 32 + k4 + 2] = (bh)xa.z;  Ab[rr * 32 + k4 + 3] = (bh)xa.w;
      const float4 wa =
          *(const float4*)&Wbase[(size_t)(e0 + rr) * D_ + kbase + k4];
      Bb[rr * 32 + k4 + 0] = (bh)wa.x;  Bb[rr * 32 + k4 + 1] = (bh)wa.y;
      Bb[rr * 32 + k4 + 2] = (bh)wa.z;  Bb[rr * 32 + k4 + 3] = (bh)wa.w;
    }
    __syncthreads();

    if (kb + 1 < D_ / 32) {                  // prefetch next K-chunk (WGP scope)
      __builtin_prefetch(&X[(size_t)(m0 + (tid & 63)) * D_ + kbase + 32], 0, 3);
      __builtin_prefetch(&Wbase[(size_t)(e0 + (tid & 63)) * D_ + kbase + 32], 0, 3);
    }

    v16bf a;
#pragma unroll
    for (int e = 0; e < 16; ++e)
      a[e] = Ab[(wave * 16 + ln) * 32 + khalf + e];
#pragma unroll
    for (int nt = 0; nt < 4; ++nt) {
      const v16bf bmf = *(const v16bf*)&Bb[(nt * 16 + ln) * 32 + lh * 16];
      acc[nt] = __builtin_amdgcn_wmma_f32_16x16x32_bf16(false, a, false, bmf,
                                                        (short)0, acc[nt],
                                                        false, false);
    }
    __syncthreads();
  }
#pragma unroll
  for (int nt = 0; nt < 4; ++nt)
#pragma unroll
    for (int r = 0; r < 8; ++r)
      QKV[(size_t)(m0 + wave * 16 + r + 8 * lh) * ETOT_ +
          n0 + nt * 16 + ln] = acc[nt][r];
}

// ---------------------------------------------------------------------------
// Kernel 2: RMS-norm + RoPE. One block per (b,s); emits bf16:
//   Qb (B,H,S,HD), Kb (B,KV,S,HD), KT (B,KV,HD,S), VT (B,KV,HD,S)
// ---------------------------------------------------------------------------
__global__ void __launch_bounds__(128) k_normrope(const float* __restrict__ QKV,
                                                  const float* __restrict__ cosb,
                                                  const float* __restrict__ sinb,
                                                  const float* __restrict__ qw,
                                                  const float* __restrict__ kw,
                                                  bh* __restrict__ Qb,
                                                  bh* __restrict__ Kb,
                                                  bh* __restrict__ KTb,
                                                  bh* __restrict__ VTb)
{
  __shared__ float vec[HD_];
  __shared__ float red[HD_];
  const int tid = threadIdx.x;                   // = hd index
  const int bs  = blockIdx.x;
  const int b   = bs >> 10;
  const int s   = bs & 1023;
  const size_t rowoff = (size_t)bs * ETOT_;
  const float c  = cosb[(size_t)bs * HD_ + tid];
  const float sn = sinb[(size_t)bs * HD_ + tid];

  for (int task = 0; task < H_ + 2 * KV_; ++task) {
    float val;
    if (task < H_)            val = QKV[rowoff + task * HD_ + tid];
    else if (task < H_ + KV_) val = QKV[rowoff + EQ_ + (task - H_) * HD_ + tid];
    else                      val = QKV[rowoff + EQ_ + EK_ + (task - H_ - KV_) * HD_ + tid];

    if (task < H_ + KV_) {
      red[tid] = val * val;
      __syncthreads();
      for (int o = 64; o > 0; o >>= 1) {
        if (tid < o) red[tid] += red[tid + o];
        __syncthreads();
      }
      const float inv = rsqrtf(red[0] / (float)HD_ + 1e-6f);
      const float w = (task < H_) ? qw[tid] : kw[tid];
      const float xn = val * inv * w;
      vec[tid] = xn;
      __syncthreads();
      const float other = (tid < HD_/2) ? -vec[tid + HD_/2] : vec[tid - HD_/2];
      const float out = xn * c + other * sn;
      __syncthreads();                           // before vec/red reuse
      if (task < H_) {
        Qb[(((size_t)(b * H_ + task)) * S_ + s) * HD_ + tid] = (bh)out;
      } else {
        const int j = task - H_;
        Kb [(((size_t)(b * KV_ + j)) * S_ + s) * HD_ + tid] = (bh)out;
        KTb[(((size_t)(b * KV_ + j)) * HD_ + tid) * S_ + s] = (bh)out;
      }
    } else {
      const int j = task - H_ - KV_;
      VTb[(((size_t)(b * KV_ + j)) * HD_ + tid) * S_ + s] = (bh)val;
    }
  }
}

// ---------------------------------------------------------------------------
// Kernel 3: 3-step Hopfield attention. One block = 16 query rows of one (b,h).
// Dynamic LDS: scores 16x1024 f32 | state 16x128 bf16 @ byte 65536 | stats.
// ---------------------------------------------------------------------------
__global__ void __launch_bounds__(128) k_attn(const bh* __restrict__ Qb,
                                              const bh* __restrict__ Kb,
                                              const bh* __restrict__ KTb,
                                              const bh* __restrict__ VTb,
                                              const float* __restrict__ log_beta,
                                              bh* __restrict__ ctx,
                                              float* __restrict__ attn_out)
{
  extern __shared__ char smem[];
  float* sc   = (float*)smem;                              // 16*1024 f32
  bh*    st   = (bh*)(smem + 16 * 1024 * 4);               // 16*128 bf16
  float* rmax = (float*)(smem + 16 * 1024 * 4 + 16 * 128 * 2);
  float* rsum = rmax + 16 * 8;

  const int tid  = threadIdx.x;
  const int wave = tid >> 5;
  const int lane = tid & 31;
  const int lh = lane >> 4, ln = lane & 15;

  int idx = blockIdx.x;
  const int qb = idx & 63;  idx >>= 6;
  const int h  = idx & 15;  idx >>= 4;
  const int b  = idx;
  const int qbase = qb * 16;
  const int kvh = h >> 1;                                  // N_REP = 2

  const float scale = SCALING_ * expf(log_beta[0]);

  // initial state = rope'd q : async global -> LDS copy (CDNA5 ASYNCcnt path)
  {
    const bh* qsrc = Qb + (((size_t)(b * H_ + h)) * S_ + qbase) * HD_;
    const unsigned long long ga = (unsigned long long)qsrc;
#pragma unroll
    for (int i = 0; i < 2; ++i) {
      const unsigned c = (unsigned)tid + i * 128u;         // 256 x 16B chunks
      const unsigned lds_addr = 65536u + c * 16u;          // st's LDS offset
      const unsigned long long g = ga + (unsigned long long)c * 16u;
      asm volatile("global_load_async_to_lds_b128 %0, %1, off"
                   :: "v"(lds_addr), "v"(g) : "memory");
    }
    asm volatile("s_wait_asynccnt 0x0" ::: "memory");
  }
  __syncthreads();

  const bh* Krow = Kb  + ((size_t)(b * KV_ + kvh)) * S_ * HD_;
  const bh* KT   = KTb + ((size_t)(b * KV_ + kvh)) * HD_ * S_;
  const bh* VT   = VTb + ((size_t)(b * KV_ + kvh)) * HD_ * S_;
  const int khalf = lh * 16;

  for (int step = 0; step < 3; ++step) {
    // ---- phase 1: scores = state @ K^T (causal tiles only) ----
    // A-fragments depend only on kk -> hoist out of the key-tile loop.
    v16bf afr[4];
#pragma unroll
    for (int kk = 0; kk < 4; ++kk)
#pragma unroll
      for (int e = 0; e < 16; ++e)
        afr[kk][e] = st[ln * 128 + kk * 32 + khalf + e];

    for (int ct = wave; ct < 64; ct += 4) {
      if (ct > qb) {
#pragma unroll
        for (int r = 0; r < 8; ++r)
          sc[(r + 8 * lh) * 1024 + ct * 16 + ln] = NEGBIG_;
      } else {
        v8f acc = {};
        const bh* kptr = Krow + (size_t)(ct * 16 + ln) * HD_;
#pragma unroll
        for (int kk = 0; kk < 4; ++kk) {
          const v16bf bmf = *(const v16bf*)(kptr + kk * 32 + lh * 16);
          acc = __builtin_amdgcn_wmma_f32_16x16x32_bf16(false, afr[kk], false,
                                                        bmf, (short)0, acc,
                                                        false, false);
        }
#pragma unroll
        for (int r = 0; r < 8; ++r) {
          const int m = r + 8 * lh, col = ct * 16 + ln;
          sc[m * 1024 + col] =
              acc[r] * scale + ((col <= qbase + m) ? 0.0f : NEGBIG_);
        }
      }
    }
    __syncthreads();

    // ---- softmax over each of the 16 rows (8 threads per row) ----
    {
      const int r = tid >> 3, seg = tid & 7;
      const int skew = ((seg * 16 + r) & 127);             // LDS bank stagger
      float* row = sc + r * 1024 + seg * 128;
      float lmax = -3.0e38f;
      for (int i = 0; i < 128; ++i) lmax = fmaxf(lmax, row[(i + skew) & 127]);
      rmax[r * 8 + seg] = lmax;
      __syncthreads();
      float mx = rmax[r * 8];
#pragma unroll
      for (int i = 1; i < 8; ++i) mx = fmaxf(mx, rmax[r * 8 + i]);
      float lsum = 0.f;
      for (int i = 0; i < 128; ++i) {
        const int ii = (i + skew) & 127;
        const float e2 = __expf(row[ii] - mx);
        row[ii] = e2;
        lsum += e2;
      }
      rsum[r * 8 + seg] = lsum;
      __syncthreads();
      float ssum = 0.f;
#pragma unroll
      for (int i = 0; i < 8; ++i) ssum += rsum[r * 8 + i];
      const float inv = 1.0f / ssum;
      for (int i = 0; i < 128; ++i) row[(i + skew) & 127] *= inv;
    }
    __syncthreads();

    if (step == 2) {  // dump last-step attention weights (contiguous 64KB)
      float* dst = attn_out + (((size_t)(b * H_ + h)) * S_ + qbase) * S_;
      for (int i = 0; i < 128; ++i) dst[tid + i * 128] = sc[tid + i * 128];
    }

    // ---- phase 2: state' = probs @ K  (V on last step) ----
    const bh* Bm = (step == 2) ? VT : KT;
    const int nkc = (qbase + 16 + 31) >> 5;                // causal K-chunks
#pragma unroll
    for (int t = 0; t < 2; ++t) {
      const int hdt = wave * 2 + t;                        // 8 hd-tiles/4 waves
      v8f acc = {};
      const bh* bptr = Bm + (size_t)(hdt * 16 + ln) * S_;
      for (int kc = 0; kc < nkc; ++kc) {
        v16bf a;
#pragma unroll
        for (int e = 0; e < 16; ++e)
          a[e] = (bh)sc[ln * 1024 + kc * 32 + khalf + e];
        const v16bf bmf = *(const v16bf*)(bptr + kc * 32 + lh * 16);
        acc = __builtin_amdgcn_wmma_f32_16x16x32_bf16(false, a, false, bmf,
                                                      (short)0, acc,
                                                      false, false);
      }
      if (step < 2) {
#pragma unroll
        for (int r = 0; r < 8; ++r)
          st[(r + 8 * lh) * 128 + hdt * 16 + ln] = (bh)acc[r];
      } else {
#pragma unroll
        for (int r = 0; r < 8; ++r)
          ctx[(size_t)(b * S_ + qbase + r + 8 * lh) * EQ_ +
              h * HD_ + hdt * 16 + ln] = (bh)acc[r];
      }
    }
    __syncthreads();
  }
}

// ---------------------------------------------------------------------------
// Kernel 4: output projection. out(f32, 2048x2048) = ctx(bf16) @ Wo^T
// Same 64x64 LDS-staged blocking as k_qkv.
// ---------------------------------------------------------------------------
__global__ void __launch_bounds__(128) k_out(const bh* __restrict__ ctx,
                                             const float* __restrict__ Wo,
                                             float* __restrict__ out)
{
  __shared__ __align__(32) bh Ab[64 * 32];
  __shared__ __align__(32) bh Bb[64 * 32];
  const int tid  = threadIdx.x;
  const int wave = tid >> 5;
  const int lane = tid & 31;
  const int lh = lane >> 4, ln = lane & 15;
  const int bmi = blockIdx.x >> 5;               // 32 x 32 blocks
  const int bni = blockIdx.x & 31;
  const int m0 = bmi * 64, n0 = bni * 64;
  const int khalf = lh * 16;

  v8f acc[4] = {};

  for (int kb = 0; kb < EQ_ / 32; ++kb) {
    const int kbase = kb * 32;
#pragma unroll
    for (int i = 0; i < 4; ++i) {
      const int c  = tid + i * 128;
      const int rr = c >> 3;
      const int k4 = (c & 7) * 4;
      // ctx is already bf16: 8B copy
      *(uint2*)&Ab[rr * 32 + k4] =
          *(const uint2*)&ctx[(size_t)(m0 + rr) * EQ_ + kbase + k4];
      const float4 wa =
          *(const float4*)&Wo[(size_t)(n0 + rr) * EQ_ + kbase + k4];
      Bb[rr * 32 + k4 + 0] = (bh)wa.x;  Bb[rr * 32 + k4 + 1] = (bh)wa.y;
      Bb[rr * 32 + k4 + 2] = (bh)wa.z;  Bb[rr * 32 + k4 + 3] = (bh)wa.w;
    }
    __syncthreads();

    if (kb + 1 < EQ_ / 32) {                     // WGP-scope prefetch
      __builtin_prefetch(&ctx[(size_t)(m0 + (tid & 63)) * EQ_ + kbase + 32], 0, 3);
      __builtin_prefetch(&Wo [(size_t)(n0 + (tid & 63)) * EQ_ + kbase + 32], 0, 3);
    }

    v16bf a;
#pragma unroll
    for (int e = 0; e < 16; ++e)
      a[e] = Ab[(wave * 16 + ln) * 32 + khalf + e];
#pragma unroll
    for (int nt = 0; nt < 4; ++nt) {
      const v16bf bmf = *(const v16bf*)&Bb[(nt * 16 + ln) * 32 + lh * 16];
      acc[nt] = __builtin_amdgcn_wmma_f32_16x16x32_bf16(false, a, false, bmf,
                                                        (short)0, acc[nt],
                                                        false, false);
    }
    __syncthreads();
  }
#pragma unroll
  for (int nt = 0; nt < 4; ++nt)
#pragma unroll
    for (int r = 0; r < 8; ++r)
      out[(size_t)(m0 + wave * 16 + r + 8 * lh) * D_ +
          n0 + nt * 16 + ln] = acc[nt][r];
}

// ---------------------------------------------------------------------------
extern "C" void kernel_launch(void* const* d_in, const int* in_sizes, int n_in,
                              void* d_out, int out_size, void* d_ws, size_t ws_size,
                              hipStream_t stream)
{
  (void)in_sizes; (void)n_in; (void)out_size; (void)ws_size;
  const float* hidden = (const float*)d_in[0];
  const float* cosb   = (const float*)d_in[1];
  const float* sinb   = (const float*)d_in[2];
  /* d_in[3] attention_mask: pure causal, computed analytically in-kernel */
  const float* Wq = (const float*)d_in[4];
  const float* Wk = (const float*)d_in[5];
  const float* Wv = (const float*)d_in[6];
  const float* Wo = (const float*)d_in[7];
  const float* qw = (const float*)d_in[8];
  const float* kw = (const float*)d_in[9];
  const float* lb = (const float*)d_in[10];

  // workspace layout (all 256B-aligned, ~60 MB total; fits in 192 MB L2)
  char* ws = (char*)d_ws;
  size_t o = 0;
  float* QKV = (float*)(ws + o); o += (size_t)M_ * ETOT_ * 4;          // 33.5 MB
  bh* Qb  = (bh*)(ws + o); o += (size_t)B_ * H_  * S_ * HD_ * 2;       //  8.4 MB
  bh* Kb  = (bh*)(ws + o); o += (size_t)B_ * KV_ * S_ * HD_ * 2;       //  4.2 MB
  bh* KTb = (bh*)(ws + o); o += (size_t)B_ * KV_ * S_ * HD_ * 2;       //  4.2 MB
  bh* VTb = (bh*)(ws + o); o += (size_t)B_ * KV_ * S_ * HD_ * 2;       //  4.2 MB
  bh* ctx = (bh*)(ws + o); o += (size_t)M_ * EQ_ * 2;                  //  8.4 MB

  float* out  = (float*)d_out;
  float* attn = out + (size_t)B_ * S_ * D_;

  const size_t smem = 16 * 1024 * 4 + 16 * 128 * 2 + 2 * 16 * 8 * 4;   // ~70 KB
  (void)hipFuncSetAttribute((const void*)k_attn,
                            hipFuncAttributeMaxDynamicSharedMemorySize,
                            (int)smem);

  k_qkv     <<<32 * 64, 128, 0, stream>>>(hidden, Wq, Wk, Wv, QKV);
  k_normrope<<<B_ * S_, 128, 0, stream>>>(QKV, cosb, sinb, qw, kw,
                                          Qb, Kb, KTb, VTb);
  k_attn    <<<B_ * H_ * (S_ / 16), 128, smem, stream>>>(Qb, Kb, KTb, VTb, lb,
                                                         ctx, attn);
  k_out     <<<32 * 32, 128, 0, stream>>>(ctx, Wo, out);
}
